// MultiHeadSelfAttention_81956565942912
// MI455X (gfx1250) — matrix operand order repaired and checked by reading, not compiled, not run on hardware
//
#include <hip/hip_runtime.h>
#include <hip/hip_bf16.h>

typedef __attribute__((ext_vector_type(16))) __bf16        v16bf;
typedef __attribute__((ext_vector_type(8)))  float         v8f;
typedef __attribute__((ext_vector_type(8)))  unsigned int  v8u;
typedef __attribute__((ext_vector_type(4)))  int           v4i;

#define EMBED 768
#define HEADS 12
#define HD 64
#define SEQ 196
#define BATCH 256
#define TOKENS (BATCH * SEQ)          // 50176
#define QKV_N (3 * EMBED)             // 2304

#if defined(__gfx1250__) && __has_builtin(__builtin_amdgcn_global_load_async_to_lds_b128)
#define ASYNC_STAGE 1
#else
#define ASYNC_STAGE 0
#endif

__device__ __forceinline__ void wait_asynccnt0() {
#if __has_builtin(__builtin_amdgcn_s_wait_asynccnt)
    __builtin_amdgcn_s_wait_asynccnt(0);
#else
    asm volatile("s_wait_asynccnt 0" ::: "memory");
#endif
}

// fp32 -> bf16 round-to-nearest-even (bit pattern)
__device__ __forceinline__ unsigned short f2bf(float f) {
    union { float f; unsigned int u; } c; c.f = f;
    unsigned int u = c.u;
    unsigned int r = u + 0x7FFFu + ((u >> 16) & 1u);
    return (unsigned short)(r >> 16);
}

__device__ __forceinline__ v8f v8f_zero() {
    v8f z = {0.f, 0.f, 0.f, 0.f, 0.f, 0.f, 0.f, 0.f};
    return z;
}

// ---------------------------------------------------------------------------
// One-time elementwise fp32 -> bf16 (n % 4 == 0, exact grid)
// ---------------------------------------------------------------------------
__global__ __launch_bounds__(256) void cvt_f32_bf16(
    const float* __restrict__ src, unsigned short* __restrict__ dst, int n)
{
    const int i = (blockIdx.x * 256 + threadIdx.x) * 4;
    if (i >= n) return;
    const float4 v = *(const float4*)&src[i];
    unsigned int lo = (unsigned int)f2bf(v.x) | ((unsigned int)f2bf(v.y) << 16);
    unsigned int hi = (unsigned int)f2bf(v.z) | ((unsigned int)f2bf(v.w) << 16);
    uint2 p; p.x = lo; p.y = hi;
    *(uint2*)&dst[i] = p;
}

// ---------------------------------------------------------------------------
// One-time W[K][N] fp32 -> Wt[N][K] bf16 (tiled transpose, K%32==0, N%32==0)
// ---------------------------------------------------------------------------
__global__ __launch_bounds__(256) void cvt_transpose_bf16(
    const float* __restrict__ W, unsigned short* __restrict__ Wt, int K, int N)
{
    __shared__ float tile[32][33];
    const int k0 = blockIdx.x * 32;
    const int n0 = blockIdx.y * 32;
    const int tx = threadIdx.x & 31;
    const int ty = threadIdx.x >> 5;   // 0..7
    #pragma unroll
    for (int r = ty; r < 32; r += 8)
        tile[r][tx] = W[(size_t)(k0 + r) * N + n0 + tx];
    __syncthreads();
    #pragma unroll
    for (int r = ty; r < 32; r += 8)
        Wt[(size_t)(n0 + r) * K + k0 + tx] = f2bf(tile[tx][r]);
}

// ---------------------------------------------------------------------------
// C[M,N] = A[M,K](bf16) @ Wt[N,K](bf16, pre-transposed) + bias[N] -> fp32
// Block tile 64(M) x 256(N), K-step 32. 256 threads = 8 waves as 2(M) x 4(N);
// each wave: 32x64 tile = 2x4 v_wmma_f32_16x16x32_bf16.
// Requires M%64==0, N%256==0, K%32==0; all element offsets < 2^31.
// ---------------------------------------------------------------------------
#define BM 64
#define BN 256
#define BK 32
#define LDT 40   // LDS row stride (bf16 elems): 80B -> 16B-aligned, bank-clean

__global__ __launch_bounds__(256) void gemm_bias_bf16wmma(
    const unsigned short* __restrict__ A,    // [M][K] bf16
    const unsigned short* __restrict__ Bt,   // [N][K] bf16
    const float* __restrict__ bias, float* __restrict__ C,
    int M, int N, int K)
{
    __shared__ unsigned short At[BM * LDT];   // [m][k]
    __shared__ unsigned short Bs[BN * LDT];   // [n][k]

    const int tid  = threadIdx.x;
    const int lane = tid & 31;
    const int w    = tid >> 5;     // wave 0..7
    const int wm   = w >> 2;       // 0..1  -> 32-row band
    const int wn   = w & 3;        // 0..3  -> 64-col band
    const int half = lane >> 4;
    const int lr   = lane & 15;

    const int bm0 = blockIdx.y * BM;
    const int bn0 = blockIdx.x * BN;

    // A loader: 64x32 = 2048 bf16 / 256 thr = 8 each (one b128)
    const int arow = tid >> 2;            // 0..63
    const int ak0  = (tid & 3) << 3;      // 0,8,16,24
    const unsigned aoff = (unsigned)(bm0 + arow) * (unsigned)K + (unsigned)ak0;
    // B loader: 256x32 = 8192 bf16 / 256 thr = 32 each (four b128), n = tid
    const unsigned boff = (unsigned)(bn0 + tid) * (unsigned)K;
    unsigned short* lb_base = &Bs[tid * LDT];
    unsigned short* la_base = &At[arow * LDT + ak0];

    v8f acc[2][4];
    #pragma unroll
    for (int mt = 0; mt < 2; ++mt)
        #pragma unroll
        for (int nt = 0; nt < 4; ++nt) acc[mt][nt] = v8f_zero();

    for (int kk = 0; kk < K; kk += BK) {
#if ASYNC_STAGE
        {
            // CDNA5 async global->LDS copies (ASYNCcnt-tracked, no VGPR bounce)
            __builtin_amdgcn_global_load_async_to_lds_b128(
                (__attribute__((address_space(1))) v4i*)&A[aoff + (unsigned)kk],
                (__attribute__((address_space(3))) v4i*)la_base, 0, 0);
            const unsigned short* gb = &Bt[boff + (unsigned)kk];
            #pragma unroll
            for (int i = 0; i < 4; ++i)
                __builtin_amdgcn_global_load_async_to_lds_b128(
                    (__attribute__((address_space(1))) v4i*)&gb[i * 8],
                    (__attribute__((address_space(3))) v4i*)&lb_base[i * 8], 0, 0);
            wait_asynccnt0();
        }
#else
        *(uint4*)la_base = *(const uint4*)&A[aoff + (unsigned)kk];
        {
            const unsigned short* gb = &Bt[boff + (unsigned)kk];
            #pragma unroll
            for (int i = 0; i < 4; ++i)
                *(uint4*)&lb_base[i * 8] = *(const uint4*)&gb[i * 8];
        }
#endif
        __syncthreads();

        // fragment fill per ISA 7.12.2 (16-bit A 16x32, B 32x16)
        v8u au[2], bu[4];
        #pragma unroll
        for (int mt = 0; mt < 2; ++mt) {
            const int m = wm * 32 + mt * 16 + lr;
            #pragma unroll
            for (int j = 0; j < 8; ++j) {
                const int k = (j >> 2) * 16 + half * 8 + (j & 3) * 2;
                au[mt][j] = *(const unsigned int*)&At[m * LDT + k];
            }
        }
        #pragma unroll
        for (int nt = 0; nt < 4; ++nt) {
            const int n = wn * 64 + nt * 16 + lr;
            #pragma unroll
            for (int j = 0; j < 8; ++j) {
                const int k = (j >> 2) * 16 + half * 8 + (j & 3) * 2;
                bu[nt][j] = *(const unsigned int*)&Bs[n * LDT + k];
            }
        }

        #pragma unroll
        for (int mt = 0; mt < 2; ++mt) {
            #pragma unroll
            for (int nt = 0; nt < 4; ++nt) {
                acc[mt][nt] = __builtin_amdgcn_wmma_f32_16x16x32_bf16(
                    false, __builtin_bit_cast(v16bf, au[mt]),
                    false, __builtin_bit_cast(v16bf, bu[nt]),
                    (short)0, acc[mt][nt], false, false);
            }
        }
        __syncthreads();
    }

    // epilogue: C/D f32 layout — VGPR i: M = i + 8*half, N = lane%16
    // all offsets fit in 31 bits -> pure 32-bit address math
    #pragma unroll
    for (int mt = 0; mt < 2; ++mt) {
        #pragma unroll
        for (int nt = 0; nt < 4; ++nt) {
            const unsigned col = (unsigned)(bn0 + wn * 64 + nt * 16 + lr);
            const float b = bias[col];
            const unsigned base =
                (unsigned)(bm0 + wm * 32 + mt * 16 + half * 8) * (unsigned)N + col;
            #pragma unroll
            for (int i = 0; i < 8; ++i)
                C[base + (unsigned)i * (unsigned)N] = acc[mt][nt][i] + b;
        }
    }
}

// ---------------------------------------------------------------------------
// Per-token attention over the HEADS axis (12x12), faithful to the reference.
// One 128-thread block per token; output written directly as bf16 (it is the
// A-matrix of the next GEMM).
// ---------------------------------------------------------------------------
__global__ __launch_bounds__(128) void attn_kernel(
    const float* __restrict__ QKV, unsigned short* __restrict__ OUT)
{
    __shared__ float row[QKV_N];
    __shared__ float sc[HEADS * HEADS];

    const int r   = blockIdx.x;
    const int tid = threadIdx.x;
    const float* src = QKV + (size_t)r * QKV_N;

    for (int i = tid; i < QKV_N; i += 128) row[i] = src[i];
    __syncthreads();

    for (int idx = tid; idx < HEADS * HEADS; idx += 128) {
        const int i = idx / HEADS, j = idx % HEADS;
        const float* q = &row[i * 192];
        const float* k = &row[j * 192 + 64];
        float s = 0.f;
        #pragma unroll
        for (int d = 0; d < HD; ++d) s += q[d] * k[d];
        sc[idx] = s * 0.125f;   // 1/sqrt(64)
    }
    __syncthreads();

    if (tid < HEADS) {
        float mx = -3.402823466e38f;
        #pragma unroll
        for (int j = 0; j < HEADS; ++j) mx = fmaxf(mx, sc[tid * HEADS + j]);
        float e[HEADS], sum = 0.f;
        #pragma unroll
        for (int j = 0; j < HEADS; ++j) { e[j] = __expf(sc[tid * HEADS + j] - mx); sum += e[j]; }
        const float inv = 1.f / sum;
        #pragma unroll
        for (int j = 0; j < HEADS; ++j) sc[tid * HEADS + j] = e[j] * inv;
    }
    __syncthreads();

    unsigned short* dst = OUT + (size_t)r * EMBED;
    for (int idx = tid; idx < EMBED; idx += 128) {
        const int i = idx >> 6, d = idx & 63;
        float o = 0.f;
        #pragma unroll
        for (int j = 0; j < HEADS; ++j) o += sc[i * HEADS + j] * row[j * 192 + 128 + d];
        dst[idx] = f2bf(o);   // idx == i*64 + d
    }
}

// ---------------------------------------------------------------------------
// logits[b,c] = sum_j O[b, j] * cls_w[j, c] + cls_b[c];  j < 150528, c < 3
// ---------------------------------------------------------------------------
__global__ __launch_bounds__(256) void cls_kernel(
    const float* __restrict__ O, const float* __restrict__ Wc,
    const float* __restrict__ bc, float* __restrict__ out)
{
    const int b   = blockIdx.x;
    const int tid = threadIdx.x;
    const int JN  = SEQ * EMBED;   // 150528

    float a0 = 0.f, a1 = 0.f, a2 = 0.f;
    const float* x = O + (size_t)b * JN;
    for (int j = tid; j < JN; j += 256) {
        const float xv = x[j];
        const float* wr = Wc + (size_t)j * 3;
        a0 += xv * wr[0]; a1 += xv * wr[1]; a2 += xv * wr[2];
    }

    __shared__ float red[3][256];
    red[0][tid] = a0; red[1][tid] = a1; red[2][tid] = a2;
    __syncthreads();
    for (int s = 128; s > 0; s >>= 1) {
        if (tid < s) {
            red[0][tid] += red[0][tid + s];
            red[1][tid] += red[1][tid + s];
            red[2][tid] += red[2][tid + s];
        }
        __syncthreads();
    }
    if (tid < 3) out[b * 3 + tid] = red[tid][0] + bc[tid];
}

extern "C" void kernel_launch(void* const* d_in, const int* in_sizes, int n_in,
                              void* d_out, int out_size, void* d_ws, size_t ws_size,
                              hipStream_t stream) {
    const float* x      = (const float*)d_in[0];
    const float* qkv_w  = (const float*)d_in[1];
    const float* qkv_b  = (const float*)d_in[2];
    const float* out_w  = (const float*)d_in[3];
    const float* out_b  = (const float*)d_in[4];
    const float* cls_w  = (const float*)d_in[5];
    const float* cls_b  = (const float*)d_in[6];
    float* out = (float*)d_out;

    // workspace partition (all sizes multiples of 16B)
    char* ws = (char*)d_ws;
    float*          QKV    = (float*)ws;                         // 50176x2304 f32
    ws += (size_t)TOKENS * QKV_N * sizeof(float);
    unsigned short* Xbf    = (unsigned short*)ws;                // 50176x768 bf16
    ws += (size_t)TOKENS * EMBED * sizeof(unsigned short);
    unsigned short* WqkvT  = (unsigned short*)ws;                // 2304x768 bf16
    ws += (size_t)QKV_N * EMBED * sizeof(unsigned short);
    unsigned short* WoutT  = (unsigned short*)ws;                // 768x768 bf16
    ws += (size_t)EMBED * EMBED * sizeof(unsigned short);
    unsigned short* ATTbf  = (unsigned short*)ws;                // 50176x768 bf16
    ws += (size_t)TOKENS * EMBED * sizeof(unsigned short);
    float*          O      = (float*)ws;                         // 50176x768 f32

    // 0) one-time conversions
    {
        const int n = TOKENS * EMBED;                            // 38,535,168
        cvt_f32_bf16<<<n / (256 * 4), 256, 0, stream>>>(x, Xbf, n);
        dim3 g1(EMBED / 32, QKV_N / 32);                         // (24, 72)
        cvt_transpose_bf16<<<g1, 256, 0, stream>>>(qkv_w, WqkvT, EMBED, QKV_N);
        dim3 g2(EMBED / 32, EMBED / 32);                         // (24, 24)
        cvt_transpose_bf16<<<g2, 256, 0, stream>>>(out_w, WoutT, EMBED, EMBED);
    }
    // 1) QKV projection: (50176 x 768) @ (768 x 2304) + b
    {
        dim3 grid(QKV_N / BN, TOKENS / BM);                      // (9, 784)
        gemm_bias_bf16wmma<<<grid, 256, 0, stream>>>(Xbf, WqkvT, qkv_b, QKV,
                                                     TOKENS, QKV_N, EMBED);
    }
    // 2) per-token 12x12 heads-axis attention (bf16 output)
    attn_kernel<<<TOKENS, 128, 0, stream>>>(QKV, ATTbf);

    // 3) output projection: (50176 x 768) @ (768 x 768) + b
    {
        dim3 grid(EMBED / BN, TOKENS / BM);                      // (3, 784)
        gemm_bias_bf16wmma<<<grid, 256, 0, stream>>>(ATTbf, WoutT, out_b, O,
                                                     TOKENS, EMBED, EMBED);
    }
    // 4) classifier head
    cls_kernel<<<BATCH, 256, 0, stream>>>(O, cls_w, cls_b, out);
}
